// TransformerQEC_58471684767985
// MI455X (gfx1250) — compile-verified, matrix operand order checked by reading, not run on hardware
//
#include <hip/hip_runtime.h>
#include <hip/hip_bf16.h>

// ---------------- types for CDNA5 WMMA ----------------
typedef __bf16 bf16;
typedef __attribute__((ext_vector_type(16))) __bf16 v16bf;
typedef __attribute__((ext_vector_type(8)))  float  v8f;

// ---------------- problem dims ----------------
#define BB    16
#define SEQ   1024
#define S1    1025          // with CLS
#define S1P   1056          // padded to multiple of 32
#define DD    512
#define HH    8
#define HDD   64
#define LL    6
#define FFND  2048
#define MROWS (BB * S1P)    // 16896
#define MTILES (MROWS / 16) // 1056
#define ST16  (S1P / 16)    // 66 key/query 16-tiles
#define ST32  (S1P / 32)    // 33 key 32-chunks

// ---------------- WMMA operand swizzle (ISA 7.12.2, 16-bit A/B tiles) ----------------
// For a 16(row/col mn) x 32(reduction kl) tile stored as [32 lanes][16 elems]:
//   lane = (mn & 15) + 16 * ((kl >> 3) & 1)
//   elem = 2 * (((kl >> 4) << 2) | ((kl & 6) >> 1)) + (kl & 1)
// Then the operand fetch is one contiguous v16bf (32B) per lane.
__device__ __forceinline__ int swz_off(int mn, int kl) {
  int half = (kl >> 3) & 1;
  int slot = ((kl >> 4) << 2) | ((kl & 6) >> 1);
  return ((mn & 15) + 16 * half) * 16 + 2 * slot + (kl & 1);
}

__device__ __forceinline__ float gelu_tanh(float x) {
  float x3 = x * x * x;
  float t = tanhf(0.7978845608028654f * (x + 0.044715f * x3));
  return 0.5f * x * (1.0f + t);
}

// ---------------- fp32 weights -> bf16 in B-operand swizzle ----------------
// dst tile id = (k>>5)*(N/16) + (n>>4); within tile: swz_off(n, k&31)
__global__ void cvt_w_swz_n(const float* __restrict__ src, bf16* __restrict__ dst,
                            int n, int nshift) {
  int N = 1 << nshift;
  int ntiles = N >> 4;
  for (int idx = blockIdx.x * blockDim.x + threadIdx.x; idx < n;
       idx += gridDim.x * blockDim.x) {
    int k = idx >> nshift;
    int c = idx & (N - 1);
    size_t tile = (size_t)(k >> 5) * ntiles + (c >> 4);
    dst[tile * 512 + swz_off(c, k & 31)] = (bf16)src[idx];
  }
}

// ---------------- embedding: p_error MLP -> pv[B,D] ----------------
__global__ void embed_pvec(const float* __restrict__ pe, const float* __restrict__ Wp1,
                           const float* __restrict__ bp1, const float* __restrict__ Wp2,
                           const float* __restrict__ bp2, float* __restrict__ pv) {
  int b = blockIdx.x, t = threadIdx.x;
  __shared__ float g[DD];
  float p = pe[b];
  for (int e = t; e < DD; e += blockDim.x) g[e] = gelu_tanh(p * Wp1[e] + bp1[e]);
  __syncthreads();
  for (int d = t; d < DD; d += blockDim.x) {
    float acc = bp2[d];
    for (int e = 0; e < DD; ++e) acc += g[e] * Wp2[e * DD + d];
    pv[b * DD + d] = acc;
  }
}

// ---------------- embedding: fill x[B,S1P,D] (padded rows zeroed) ----------------
__global__ void embed_fill(const int* __restrict__ syn, const float* __restrict__ W_in,
                           const float* __restrict__ b_in, const float* __restrict__ cls,
                           const float* __restrict__ pv, float* __restrict__ X) {
  int row = blockIdx.x;
  int b = row / S1P, s = row % S1P, t = threadIdx.x;
  float* xr = X + (size_t)row * DD;
  const float* pvb = pv + b * DD;
  if (s == 0) {
    for (int d = t; d < DD; d += blockDim.x) xr[d] = cls[d] + pvb[d];
  } else if (s <= SEQ) {
    float sv = (float)syn[b * SEQ + (s - 1)];
    for (int d = t; d < DD; d += blockDim.x) xr[d] = sv * W_in[d] + b_in[d] + pvb[d];
  } else {
    for (int d = t; d < DD; d += blockDim.x) xr[d] = 0.0f;
  }
}

// ---------------- LayerNorm (fp32 in -> bf16 A-operand-swizzled out) ----------------
__global__ void ln_bf16_swz(const float* __restrict__ X, const float* __restrict__ sc,
                            const float* __restrict__ bi, bf16* __restrict__ Y) {
  int row = blockIdx.x, t = threadIdx.x;
  const float* xr = X + (size_t)row * DD;
  __shared__ float red[256];
  float a0 = xr[t], a1 = xr[t + 256];
  red[t] = a0 + a1;
  __syncthreads();
  for (int o = 128; o > 0; o >>= 1) { if (t < o) red[t] += red[t + o]; __syncthreads(); }
  float mean = red[0] * (1.0f / DD);
  __syncthreads();
  float d0 = a0 - mean, d1 = a1 - mean;
  red[t] = d0 * d0 + d1 * d1;
  __syncthreads();
  for (int o = 128; o > 0; o >>= 1) { if (t < o) red[t] += red[t + o]; __syncthreads(); }
  float rstd = rsqrtf(red[0] * (1.0f / DD) + 1e-6f);
  float y0 = d0 * rstd * sc[t] + bi[t];
  float y1 = d1 * rstd * sc[t + 256] + bi[t + 256];
  int mt = row >> 4, m = row & 15;
  size_t base = (size_t)mt * (DD / 32) * 512;
  int c0 = t, c1 = t + 256;
  Y[base + (size_t)(c0 >> 5) * 512 + swz_off(m, c0 & 31)] = (bf16)y0;
  Y[base + (size_t)(c1 >> 5) * 512 + swz_off(m, c1 & 31)] = (bf16)y1;
}

// ---------------- WMMA GEMM, no LDS, register-blocked 16x64 per wave ----------------
// A: [mt][K/32 tiles][512], W: [kt][N/16 tiles][512]. 4 waves/block -> 16x256 of N.
// A operand loaded once per K-step and reused across 4 WMMAs (4 B tiles).
__global__ __launch_bounds__(128) void gemm_wmma(
    const bf16* __restrict__ Aswz, const bf16* __restrict__ Wswz,
    const float* __restrict__ bias, const float* __restrict__ R,
    float* __restrict__ Of, bf16* __restrict__ Ob, bf16* __restrict__ Oswz,
    int N, int K, int act) {
  const int mt = blockIdx.x;
  const int wave = threadIdx.x >> 5;
  const int nt0 = (blockIdx.y * 4 + wave) * 4;   // first of 4 consecutive 16-col tiles
  const int lane = threadIdx.x & 31;
  const int half = lane >> 4;
  const int id16 = lane & 15;
  const int kt_n = K >> 5;
  const int nt_n = N >> 4;

  const bf16* Ab = Aswz + (size_t)mt * kt_n * 512 + lane * 16;
  const bf16* Wb = Wswz + (size_t)nt0 * 512 + lane * 16;

  v8f acc0 = {0.f, 0.f, 0.f, 0.f, 0.f, 0.f, 0.f, 0.f};
  v8f acc1 = acc0, acc2 = acc0, acc3 = acc0;
#pragma unroll 2
  for (int kt = 0; kt < kt_n; ++kt) {
    v16bf av = *(const v16bf*)(Ab + (size_t)kt * 512);
    const bf16* wp = Wb + (size_t)kt * nt_n * 512;
    v16bf b0 = *(const v16bf*)(wp);
    v16bf b1 = *(const v16bf*)(wp + 512);
    v16bf b2 = *(const v16bf*)(wp + 1024);
    v16bf b3 = *(const v16bf*)(wp + 1536);
    acc0 = __builtin_amdgcn_wmma_f32_16x16x32_bf16(false, av, false, b0, (short)0, acc0, false, false);
    acc1 = __builtin_amdgcn_wmma_f32_16x16x32_bf16(false, av, false, b1, (short)0, acc1, false, false);
    acc2 = __builtin_amdgcn_wmma_f32_16x16x32_bf16(false, av, false, b2, (short)0, acc2, false, false);
    acc3 = __builtin_amdgcn_wmma_f32_16x16x32_bf16(false, av, false, b3, (short)0, acc3, false, false);
  }
#pragma unroll
  for (int t = 0; t < 4; ++t) {
    v8f acc = (t == 0) ? acc0 : (t == 1) ? acc1 : (t == 2) ? acc2 : acc3;
    int oc = (nt0 + t) * 16 + id16;
#pragma unroll
    for (int r = 0; r < 8; ++r) {
      int m = r + 8 * half;
      size_t orow = (size_t)(mt * 16 + m);
      float v = acc[r] + (bias ? bias[oc] : 0.0f);
      if (act) v = gelu_tanh(v);
      if (R) v += R[orow * N + oc];
      if (Of) Of[orow * N + oc] = v;
      if (Ob) Ob[orow * N + oc] = (bf16)v;
      if (Oswz) Oswz[((size_t)mt * (N >> 5) + (oc >> 5)) * 512 + swz_off(m, oc & 31)] = (bf16)v;
    }
  }
}

// ---------------- RoPE + swizzle for attention operands ----------------
// Q and K share the same layout: per (b,h): [66 16-tiles][2 feat-halves][512]
__global__ void rope_qk_swz(const bf16* __restrict__ src, bf16* __restrict__ dst,
                            const float* __restrict__ cosT, const float* __restrict__ sinT,
                            int do_rope) {
  int row = blockIdx.x;
  int b = row / S1P, s = row % S1P;
  int t = threadIdx.x;            // 256 = 8 heads * 32 pairs
  int h = t >> 5, e = t & 31;
  int bh = b * HH + h;
  const bf16* sr = src + (size_t)row * DD + h * HDD;
  float x1 = (float)sr[e], x2 = (float)sr[e + 32];
  float r1 = x1, r2 = x2;
  if (do_rope) {
    float c = 1.0f, sn = 0.0f;
    if (s >= 1 && s <= SEQ) { c = cosT[(s - 1) * 32 + e]; sn = sinT[(s - 1) * 32 + e]; }
    r1 = x1 * c - x2 * sn;
    r2 = x1 * sn + x2 * c;
  }
  size_t tb = ((size_t)bh * ST16 + (s >> 4)) * 2;
  dst[(tb + 0) * 512 + swz_off(s, e)] = (bf16)r1;   // feat e    in half 0
  dst[(tb + 1) * 512 + swz_off(s, e)] = (bf16)r2;   // feat e+32 in half 1
}

// V layout: per (b,h): [33 key-32-chunks][4 feat-16-tiles][512] (B-operand, mn=feat, kl=key)
__global__ void v_swz(const bf16* __restrict__ src, bf16* __restrict__ dst) {
  int row = blockIdx.x;
  int b = row / S1P, s = row % S1P;
  int t = threadIdx.x;
  int h = t >> 5, e = t & 31;
  int bh = b * HH + h;
  const bf16* sr = src + (size_t)row * DD + h * HDD;
  bf16 x1 = sr[e], x2 = sr[e + 32];
  int j = s >> 5, kl = s & 31;
  size_t cb = ((size_t)bh * ST32 + j) * 4;
  int e2 = e + 32;
  dst[(cb + (e >> 4)) * 512 + swz_off(e, kl)]   = x1;
  dst[(cb + (e2 >> 4)) * 512 + swz_off(e2, kl)] = x2;
}

// ---------------- flash attention: 1 wave per (b,h, 16-query tile) ----------------
__global__ __launch_bounds__(32) void flash_attn(const bf16* __restrict__ Qs,
                                                 const bf16* __restrict__ Ks,
                                                 const bf16* __restrict__ Vs,
                                                 bf16* __restrict__ ATswz) {
  const int bh = blockIdx.x;
  const int b = bh / HH, h = bh % HH;
  const int qt = blockIdx.y;
  const int lane = threadIdx.x;
  const int half = lane >> 4, id16 = lane & 15;

  __shared__ bf16 lP[512];  // P tile in A-operand swizzle

  const bf16* Qb = Qs + ((size_t)bh * ST16 + qt) * 2 * 512 + lane * 16;
  v16bf aq0 = *(const v16bf*)(Qb);
  v16bf aq1 = *(const v16bf*)(Qb + 512);

  v8f o0 = {0.f,0.f,0.f,0.f,0.f,0.f,0.f,0.f};
  v8f o1 = o0, o2 = o0, o3 = o0;
  float mrun[8], lrun[8];
#pragma unroll
  for (int r = 0; r < 8; ++r) { mrun[r] = -1e30f; lrun[r] = 0.0f; }
  const float sm_scale = 0.125f;  // 1/sqrt(64)

  const bf16* Kb = Ks + (size_t)bh * ST16 * 2 * 512 + lane * 16;
  const bf16* Vb = Vs + (size_t)bh * ST32 * 4 * 512 + lane * 16;

  for (int j = 0; j < ST32; ++j) {
    // S = Q @ K^T : two 16(key)-tiles x two feature halves
    const bf16* k0p = Kb + (size_t)(j * 2) * 2 * 512;
    const bf16* k1p = Kb + (size_t)(j * 2 + 1) * 2 * 512;
    v8f sc0 = {0.f,0.f,0.f,0.f,0.f,0.f,0.f,0.f};
    v8f sc1 = sc0;
    sc0 = __builtin_amdgcn_wmma_f32_16x16x32_bf16(false, aq0, false, *(const v16bf*)(k0p),
                                                  (short)0, sc0, false, false);
    sc0 = __builtin_amdgcn_wmma_f32_16x16x32_bf16(false, aq1, false, *(const v16bf*)(k0p + 512),
                                                  (short)0, sc0, false, false);
    sc1 = __builtin_amdgcn_wmma_f32_16x16x32_bf16(false, aq0, false, *(const v16bf*)(k1p),
                                                  (short)0, sc1, false, false);
    sc1 = __builtin_amdgcn_wmma_f32_16x16x32_bf16(false, aq1, false, *(const v16bf*)(k1p + 512),
                                                  (short)0, sc1, false, false);

    // online softmax; each row lives in one VGPR across one 16-lane half
    const int ks0 = j * 32;
    const int gk0 = ks0 + id16, gk1 = ks0 + 16 + id16;
#pragma unroll
    for (int r = 0; r < 8; ++r) {
      float v0 = (gk0 < S1) ? sc0[r] * sm_scale : -1e30f;
      float v1 = (gk1 < S1) ? sc1[r] * sm_scale : -1e30f;
      float rm = fmaxf(v0, v1);
      rm = fmaxf(rm, __shfl_xor(rm, 1, 32));
      rm = fmaxf(rm, __shfl_xor(rm, 2, 32));
      rm = fmaxf(rm, __shfl_xor(rm, 4, 32));
      rm = fmaxf(rm, __shfl_xor(rm, 8, 32));
      float mn = fmaxf(mrun[r], rm);
      float sscl = __expf(mrun[r] - mn);
      float p0 = __expf(v0 - mn), p1 = __expf(v1 - mn);
      float rs = p0 + p1;
      rs += __shfl_xor(rs, 1, 32);
      rs += __shfl_xor(rs, 2, 32);
      rs += __shfl_xor(rs, 4, 32);
      rs += __shfl_xor(rs, 8, 32);
      lrun[r] = lrun[r] * sscl + rs;
      mrun[r] = mn;
      o0[r] *= sscl; o1[r] *= sscl; o2[r] *= sscl; o3[r] *= sscl;
      int m = r + 8 * half;
      lP[swz_off(m, id16)]      = (bf16)p0;   // key kl = id16
      lP[swz_off(m, 16 + id16)] = (bf16)p1;   // key kl = 16+id16
    }
    __syncthreads();
    v16bf ap = *(const v16bf*)(&lP[lane * 16]);
    __syncthreads();

    // O += P(16x32) @ V(32x64): 4 pre-swizzled B tiles, direct global loads
    const bf16* vj = Vb + (size_t)j * 4 * 512;
    o0 = __builtin_amdgcn_wmma_f32_16x16x32_bf16(false, ap, false, *(const v16bf*)(vj),
                                                 (short)0, o0, false, false);
    o1 = __builtin_amdgcn_wmma_f32_16x16x32_bf16(false, ap, false, *(const v16bf*)(vj + 512),
                                                 (short)0, o1, false, false);
    o2 = __builtin_amdgcn_wmma_f32_16x16x32_bf16(false, ap, false, *(const v16bf*)(vj + 1024),
                                                 (short)0, o2, false, false);
    o3 = __builtin_amdgcn_wmma_f32_16x16x32_bf16(false, ap, false, *(const v16bf*)(vj + 1536),
                                                 (short)0, o3, false, false);
  }

  // store O in A-operand swizzle for the O-projection GEMM (K dim = h*64+e)
  const int mt = b * (S1P / 16) + qt;  // global 16-row tile id over [B*S1P]
#pragma unroll
  for (int r = 0; r < 8; ++r) {
    float inv = 1.0f / lrun[r];
    int m = r + 8 * half;
    float vo[4] = {o0[r] * inv, o1[r] * inv, o2[r] * inv, o3[r] * inv};
#pragma unroll
    for (int nt = 0; nt < 4; ++nt) {
      int col = h * HDD + nt * 16 + id16;
      ATswz[((size_t)mt * (DD / 32) + (col >> 5)) * 512 + swz_off(m, col & 31)] = (bf16)vo[nt];
    }
  }
}

// ---------------- classification head (tiny: plain VALU) ----------------
__global__ void head_kernel(const float* __restrict__ X, const float* __restrict__ lns,
                            const float* __restrict__ lnb, const float* __restrict__ Wh1,
                            const float* __restrict__ bh1, const float* __restrict__ Wh2,
                            const float* __restrict__ bh2, float* __restrict__ out) {
  int b = blockIdx.x, t = threadIdx.x;
  __shared__ float hn[DD];
  __shared__ float g[DD];
  __shared__ float red[256];
  const float* xr = X + (size_t)b * S1P * DD;  // CLS row
  float a0 = xr[t], a1 = xr[t + 256];
  red[t] = a0 + a1;
  __syncthreads();
  for (int o = 128; o > 0; o >>= 1) { if (t < o) red[t] += red[t + o]; __syncthreads(); }
  float mean = red[0] * (1.0f / DD);
  __syncthreads();
  float d0 = a0 - mean, d1 = a1 - mean;
  red[t] = d0 * d0 + d1 * d1;
  __syncthreads();
  for (int o = 128; o > 0; o >>= 1) { if (t < o) red[t] += red[t + o]; __syncthreads(); }
  float rstd = rsqrtf(red[0] * (1.0f / DD) + 1e-6f);
  hn[t]       = d0 * rstd * lns[t] + lnb[t];
  hn[t + 256] = d1 * rstd * lns[t + 256] + lnb[t + 256];
  __syncthreads();
  for (int c = t; c < DD; c += 256) {
    float acc = bh1[c];
    for (int e = 0; e < DD; ++e) acc += hn[e] * Wh1[e * DD + c];
    g[c] = gelu_tanh(acc);
  }
  __syncthreads();
  if (t < 2) {
    float acc = bh2[t];
    for (int e = 0; e < DD; ++e) acc += g[e] * Wh2[e * 2 + t];
    out[b * 2 + t] = acc;
  }
}

// ---------------- launcher ----------------
extern "C" void kernel_launch(void* const* d_in, const int* in_sizes, int n_in,
                              void* d_out, int out_size, void* d_ws, size_t ws_size,
                              hipStream_t stream) {
  (void)in_sizes; (void)n_in; (void)out_size; (void)ws_size;
  const int*   syn  = (const int*)d_in[0];
  const float* pe   = (const float*)d_in[1];
  const float* cosT = (const float*)d_in[3];
  const float* sinT = (const float*)d_in[4];
  const float* W_in = (const float*)d_in[5];
  const float* b_in = (const float*)d_in[6];
  const float* cls  = (const float*)d_in[7];
  const float* Wp1  = (const float*)d_in[8];
  const float* bp1  = (const float*)d_in[9];
  const float* Wp2  = (const float*)d_in[10];
  const float* bp2  = (const float*)d_in[11];
  const float* lnfs = (const float*)d_in[12];
  const float* lnfb = (const float*)d_in[13];
  const float* Wh1  = (const float*)d_in[14];
  const float* bh1  = (const float*)d_in[15];
  const float* Wh2  = (const float*)d_in[16];
  const float* bh2  = (const float*)d_in[17];
  auto LPAR = [&](int l, int k) { return (const float*)d_in[18 + 16 * l + k]; };

  char* ws = (char*)d_ws;
  size_t off = 0;
  auto alloc = [&](size_t bytes) -> void* {
    void* p = ws + off;
    off += (bytes + 255) & ~(size_t)255;
    return p;
  };
  float* X   = (float*)alloc((size_t)MROWS * DD * sizeof(float));
  bf16* YB   = (bf16*)alloc((size_t)MROWS * DD * sizeof(bf16));   // LN out, A-swizzled
  bf16* QB   = (bf16*)alloc((size_t)MROWS * DD * sizeof(bf16));   // row-major QKV
  bf16* KB   = (bf16*)alloc((size_t)MROWS * DD * sizeof(bf16));
  bf16* VB   = (bf16*)alloc((size_t)MROWS * DD * sizeof(bf16));
  bf16* Qs   = (bf16*)alloc((size_t)MROWS * DD * sizeof(bf16));   // swizzled attn operands
  bf16* Ksw  = (bf16*)alloc((size_t)MROWS * DD * sizeof(bf16));
  bf16* Vsw  = (bf16*)alloc((size_t)MROWS * DD * sizeof(bf16));
  bf16* ATs  = (bf16*)alloc((size_t)MROWS * DD * sizeof(bf16));   // attn out, A-swizzled
  bf16* FB   = (bf16*)alloc((size_t)MROWS * FFND * sizeof(bf16)); // FFN mid, A-swizzled
  bf16* WB   = (bf16*)alloc((size_t)(4 * DD * DD + 2 * DD * FFND) * sizeof(bf16));
  float* PV  = (float*)alloc((size_t)BB * DD * sizeof(float));

  embed_pvec<<<BB, 256, 0, stream>>>(pe, Wp1, bp1, Wp2, bp2, PV);
  embed_fill<<<MROWS, 256, 0, stream>>>(syn, W_in, b_in, cls, PV, X);

  const int DQ = DD * DD;
  const int DF = DD * FFND;
  bf16* wq = WB;
  bf16* wk = WB + DQ;
  bf16* wv = WB + 2 * DQ;
  bf16* wo = WB + 3 * DQ;
  bf16* w1 = WB + 4 * DQ;
  bf16* w2 = WB + 4 * DQ + DF;

  dim3 g512(MTILES, DD / 256), g2048(MTILES, FFND / 256);
  dim3 gf(BB * HH, ST16);

  for (int l = 0; l < LL; ++l) {
    cvt_w_swz_n<<<512, 256, 0, stream>>>(LPAR(l, 2), wq, DQ, 9);    // N=512
    cvt_w_swz_n<<<512, 256, 0, stream>>>(LPAR(l, 4), wk, DQ, 9);
    cvt_w_swz_n<<<512, 256, 0, stream>>>(LPAR(l, 6), wv, DQ, 9);
    cvt_w_swz_n<<<512, 256, 0, stream>>>(LPAR(l, 8), wo, DQ, 9);
    cvt_w_swz_n<<<2048, 256, 0, stream>>>(LPAR(l, 12), w1, DF, 11); // N=2048
    cvt_w_swz_n<<<2048, 256, 0, stream>>>(LPAR(l, 14), w2, DF, 9);  // N=512

    ln_bf16_swz<<<MROWS, 256, 0, stream>>>(X, LPAR(l, 0), LPAR(l, 1), YB);
    gemm_wmma<<<g512, 128, 0, stream>>>(YB, wq, LPAR(l, 3), nullptr, nullptr, QB, nullptr, DD, DD, 0);
    gemm_wmma<<<g512, 128, 0, stream>>>(YB, wk, LPAR(l, 5), nullptr, nullptr, KB, nullptr, DD, DD, 0);
    gemm_wmma<<<g512, 128, 0, stream>>>(YB, wv, LPAR(l, 7), nullptr, nullptr, VB, nullptr, DD, DD, 0);
    rope_qk_swz<<<MROWS, 256, 0, stream>>>(QB, Qs, cosT, sinT, 1);
    rope_qk_swz<<<MROWS, 256, 0, stream>>>(KB, Ksw, cosT, sinT, 1);
    v_swz<<<MROWS, 256, 0, stream>>>(VB, Vsw);
    flash_attn<<<gf, 32, 0, stream>>>(Qs, Ksw, Vsw, ATs);
    gemm_wmma<<<g512, 128, 0, stream>>>(ATs, wo, LPAR(l, 9), X, X, nullptr, nullptr, DD, DD, 0);
    ln_bf16_swz<<<MROWS, 256, 0, stream>>>(X, LPAR(l, 10), LPAR(l, 11), YB);
    gemm_wmma<<<g2048, 128, 0, stream>>>(YB, w1, LPAR(l, 13), nullptr, nullptr, nullptr, FB, FFND, DD, 1);
    gemm_wmma<<<g512, 128, 0, stream>>>(FB, w2, LPAR(l, 15), X, X, nullptr, nullptr, DD, FFND, 0);
  }

  head_kernel<<<BB, 256, 0, stream>>>(X, lnfs, lnfb, Wh1, bh1, Wh2, bh2, (float*)d_out);
}
